// LightGCN_9612136808811
// MI455X (gfx1250) — compile-verified, matrix operand order checked by reading, not run on hardware
//
#include <hip/hip_runtime.h>
#include <hip/hip_bf16.h>

#define USER_NUM   200000
#define ITEM_NUM   100000
#define CH         64
#define N_NODES    (USER_NUM + ITEM_NUM)
#define N_EDGES    4000000
#define BATCH      512
#define SEQ_LEN    50
#define NTILES     (ITEM_NUM / 16)   /* 6250 */
#define TPB        256
#define NB_SCAN    ((N_NODES + TPB - 1) / TPB)   /* 1172 */

typedef __attribute__((ext_vector_type(16))) __bf16       v16bf;
typedef __attribute__((ext_vector_type(8)))  float        v8f;
typedef __attribute__((ext_vector_type(4)))  float        v4f;
typedef __attribute__((ext_vector_type(4)))  unsigned int v4u;

union V16 { v16bf v; v4u q[2]; };

// ---------------- utility kernels ----------------
__global__ void k_zero_v4(v4u* __restrict__ p, long long n4) {
    long long i = (long long)blockIdx.x * blockDim.x + threadIdx.x;
    if (i < n4) p[i] = (v4u){0u, 0u, 0u, 0u};
}

__global__ void k_copy_v4(v4u* __restrict__ d, const v4u* __restrict__ s, long long n4) {
    long long i = (long long)blockIdx.x * blockDim.x + threadIdx.x;
    if (i < n4) d[i] = s[i];
}

__global__ void k_zero1(float* p) { p[0] = 0.0f; }

// ---------------- degree (int) + normalization ----------------
__global__ void k_deg_u32(const int* __restrict__ col, unsigned int* __restrict__ deg, int ne) {
    int e = blockIdx.x * blockDim.x + threadIdx.x;
    if (e < ne) atomicAdd(&deg[col[e]], 1u);
}

__global__ void k_dinv(const unsigned int* __restrict__ deg, float* __restrict__ dinv, int n) {
    int i = blockIdx.x * blockDim.x + threadIdx.x;
    if (i < n) {
        float d = (float)deg[i];
        dinv[i] = (d > 0.0f) ? rsqrtf(fmaxf(d, 1.0f)) : 0.0f;
    }
}

// ---------------- exclusive scan of degrees -> CSR row starts ----------------
__global__ void k_blocksum(const unsigned int* __restrict__ deg, unsigned int* __restrict__ bsum,
                           int n) {
    __shared__ unsigned int sh[TPB];
    int i = blockIdx.x * TPB + threadIdx.x;
    sh[threadIdx.x] = (i < n) ? deg[i] : 0u;
    __syncthreads();
    for (int s = TPB / 2; s > 0; s >>= 1) {
        if (threadIdx.x < s) sh[threadIdx.x] += sh[threadIdx.x + s];
        __syncthreads();
    }
    if (threadIdx.x == 0) bsum[blockIdx.x] = sh[0];
}

// Single-block sequential-chunk exclusive scan over the block sums (in place).
__global__ void k_scan_partials(unsigned int* __restrict__ bsum, int nb) {
    __shared__ unsigned int sh[TPB];
    __shared__ unsigned int carry;
    if (threadIdx.x == 0) carry = 0u;
    __syncthreads();
    for (int base = 0; base < nb; base += TPB) {
        int i = base + threadIdx.x;
        unsigned int v = (i < nb) ? bsum[i] : 0u;
        sh[threadIdx.x] = v;
        __syncthreads();
        for (int o = 1; o < TPB; o <<= 1) {
            unsigned int t = (threadIdx.x >= o) ? sh[threadIdx.x - o] : 0u;
            __syncthreads();
            sh[threadIdx.x] += t;
            __syncthreads();
        }
        unsigned int incl = sh[threadIdx.x];
        unsigned int c = carry;
        if (i < nb) bsum[i] = c + (incl - v);   // exclusive offset
        __syncthreads();
        if (threadIdx.x == TPB - 1) carry = c + incl;
        __syncthreads();
    }
}

__global__ void k_scan_final(const unsigned int* __restrict__ deg,
                             const unsigned int* __restrict__ boff,
                             unsigned int* __restrict__ start, int n) {
    __shared__ unsigned int sh[TPB];
    int i = blockIdx.x * TPB + threadIdx.x;
    unsigned int v = (i < n) ? deg[i] : 0u;
    sh[threadIdx.x] = v;
    __syncthreads();
    for (int o = 1; o < TPB; o <<= 1) {
        unsigned int t = (threadIdx.x >= o) ? sh[threadIdx.x - o] : 0u;
        __syncthreads();
        sh[threadIdx.x] += t;
        __syncthreads();
    }
    if (i < n) start[i] = boff[blockIdx.x] + (sh[threadIdx.x] - v);
}

// ---------------- CSR fill: csr_row[start[col]+k] = row ----------------
__global__ void k_fill(const int* __restrict__ row, const int* __restrict__ col,
                       unsigned int* __restrict__ cursor, int* __restrict__ csr, int ne) {
    int e = blockIdx.x * blockDim.x + threadIdx.x;
    if (e < ne) {
        unsigned int pos = atomicAdd(&cursor[col[e]], 1u);
        csr[pos] = row[e];
    }
}

// ---------------- pull-style LightGCN conv (no float atomics) ----------------
// y[c] = dinv[c] * sum_{r in in(c)} x[r] * dinv[r];  also fin += y fused.
// 16 threads per node, 4 channels (one v4f) per thread; gathers are L2-resident.
__global__ void k_conv_gather(const float* __restrict__ x, const unsigned int* __restrict__ start,
                              const unsigned int* __restrict__ deg, const int* __restrict__ csr,
                              const float* __restrict__ dinv, float* __restrict__ y,
                              float* __restrict__ fin, int n) {
    long long gid = (long long)blockIdx.x * blockDim.x + threadIdx.x;
    int node = (int)(gid >> 4);
    if (node >= n) return;
    int c = ((int)gid & 15) * 4;
    unsigned int s = start[node];
    unsigned int d = deg[node];
    float dc = dinv[node];
    v4f acc = {0.f, 0.f, 0.f, 0.f};
    for (unsigned int j = 0; j < d; ++j) {
        int r = csr[s + j];
        float w = dinv[r];
        v4f xv = *(const v4f*)(x + (size_t)r * CH + c);
        acc += xv * w;
    }
    acc *= dc;
    size_t o = (size_t)node * CH + c;
    *(v4f*)(y + o) = acc;
    v4f f = *(const v4f*)(fin + o);
    *(v4f*)(fin + o) = f + acc;
}

// ---------------- final_x*0.25 -> bf16 ----------------
__global__ void k_scale_cvt(const float* __restrict__ fin, __hip_bfloat16* __restrict__ out,
                            long long n) {
    long long i = (long long)blockIdx.x * blockDim.x + threadIdx.x;
    if (i < n) out[i] = __float2bfloat16(fin[i] * 0.25f);
}

// ---------------- gather A rows: A[m] = finb[user_idx[m]] ----------------
__global__ void k_gatherA(const __hip_bfloat16* __restrict__ finb, const int* __restrict__ uidx,
                          __hip_bfloat16* __restrict__ A, int n) {
    int i = blockIdx.x * blockDim.x + threadIdx.x;
    if (i < n) {
        int m = i / CH, c = i % CH;
        A[i] = finb[(size_t)uidx[m] * CH + c];
    }
}

// ---------------- scatter mask bytes ----------------
__global__ void k_scatter(const int* __restrict__ seq, unsigned char* __restrict__ mask, int n) {
    int i = blockIdx.x * blockDim.x + threadIdx.x;
    if (i < n) {
        int b = i / SEQ_LEN;
        mask[(size_t)b * ITEM_NUM + seq[i]] = 1;
    }
}

// ---------------- WMMA bf16 GEMM: logits[512 x 100000] = A(512x64) @ B(100000x64)^T ----
__global__ void __launch_bounds__(256)
k_gemm(const __bf16* __restrict__ A, const __bf16* __restrict__ B, float* __restrict__ out) {
    const int lane  = threadIdx.x & 31;
    const int wave  = threadIdx.x >> 5;
    const int mTile = blockIdx.y;          // 0..31
    const int ln    = lane & 15;
    const int kh    = lane >> 4;           // 0 or 1

    const __bf16* arow = A + (size_t)(mTile * 16 + ln) * CH;
    V16 a0, a1;
    a0.q[0] = *(const v4u*)(arow +      kh * 8);
    a0.q[1] = *(const v4u*)(arow + 16 + kh * 8);
    a1.q[0] = *(const v4u*)(arow + 32 + kh * 8);
    a1.q[1] = *(const v4u*)(arow + 48 + kh * 8);

    int nTileBase = (blockIdx.x * 8 + wave) * 4;
#pragma unroll
    for (int t = 0; t < 4; ++t) {
        int nTile = nTileBase + t;
        if (nTile >= NTILES) return;
        const __bf16* brow = B + (size_t)(nTile * 16 + ln) * CH;
        V16 b0, b1;
        b0.q[0] = *(const v4u*)(brow +      kh * 8);
        b0.q[1] = *(const v4u*)(brow + 16 + kh * 8);
        b1.q[0] = *(const v4u*)(brow + 32 + kh * 8);
        b1.q[1] = *(const v4u*)(brow + 48 + kh * 8);

        v8f c = {0.f, 0.f, 0.f, 0.f, 0.f, 0.f, 0.f, 0.f};
        c = __builtin_amdgcn_wmma_f32_16x16x32_bf16(false, a0.v, false, b0.v, (short)0, c,
                                                    false, false);
        c = __builtin_amdgcn_wmma_f32_16x16x32_bf16(false, a1.v, false, b1.v, (short)0, c,
                                                    false, false);

        int n  = nTile * 16 + ln;
        int mb = mTile * 16 + kh * 8;
        float* op = out + (size_t)mb * ITEM_NUM + n;
#pragma unroll
        for (int v = 0; v < 8; ++v) op[(size_t)v * ITEM_NUM] = c[v];
    }
}

// ---------------- loss: one block per batch row ----------------
__global__ void __launch_bounds__(256)
k_loss(const float* __restrict__ logits, const unsigned char* __restrict__ mask,
       float* __restrict__ loss_out) {
    int b = blockIdx.x;
    const float* lrow         = logits + (size_t)b * ITEM_NUM;
    const unsigned char* mrow = mask + (size_t)b * ITEM_NUM;
    float ps = 0.f, ns = 0.f, cnt = 0.f;
    for (int i = threadIdx.x; i < ITEM_NUM; i += TPB) {
        float z   = lrow[i];
        float t   = log1pf(__expf(-fabsf(z)));
        float lsp = fminf(z, 0.f) - t;    // log_sigmoid(z)
        float lsn = fminf(-z, 0.f) - t;   // log_sigmoid(-z)
        float m   = (float)mrow[i];
        ps += lsp * m;
        ns += lsn * (1.f - m);
        cnt += m;
    }
    __shared__ float sp[TPB], sn[TPB], sc[TPB];
    sp[threadIdx.x] = ps; sn[threadIdx.x] = ns; sc[threadIdx.x] = cnt;
    __syncthreads();
    for (int s = TPB / 2; s > 0; s >>= 1) {
        if (threadIdx.x < s) {
            sp[threadIdx.x] += sp[threadIdx.x + s];
            sn[threadIdx.x] += sn[threadIdx.x + s];
            sc[threadIdx.x] += sc[threadIdx.x + s];
        }
        __syncthreads();
    }
    if (threadIdx.x == 0) {
        float pos = sp[0] / sc[0];
        float neg = sn[0] / ((float)ITEM_NUM - sc[0]);
        unsafeAtomicAdd(loss_out, (-pos - neg) * (1.0f / (float)BATCH));
    }
}

// ---------------- host launcher ----------------
extern "C" void kernel_launch(void* const* d_in, const int* in_sizes, int n_in,
                              void* d_out, int out_size, void* d_ws, size_t ws_size,
                              hipStream_t stream) {
    const float* id_emb  = (const float*)d_in[0];
    const int*   edge    = (const int*)d_in[1];
    const int*   row     = edge;
    const int*   col     = edge + N_EDGES;
    const int*   uidx    = (const int*)d_in[2];
    const int*   seq     = (const int*)d_in[3];
    float* out    = (float*)d_out;
    float* logits = out + 1;   // d_out[0]=loss, then 512x100000 logits

    // workspace carve-up (256B aligned)
    char* base = (char*)d_ws;
    size_t off = 0;
    auto carve = [&](size_t bytes) -> char* {
        char* p = base + off;
        off = (off + bytes + 255) & ~(size_t)255;
        return p;
    };
    unsigned int*   degu   = (unsigned int*)carve((size_t)N_NODES * 4);
    unsigned int*   startp = (unsigned int*)carve((size_t)N_NODES * 4);
    unsigned int*   cursor = (unsigned int*)carve((size_t)N_NODES * 4);
    unsigned int*   bsum   = (unsigned int*)carve((size_t)NB_SCAN * 4);
    int*            csr    = (int*)carve((size_t)N_EDGES * 4);
    float*          dinv   = (float*)carve((size_t)N_NODES * 4);
    float*          xa     = (float*)carve((size_t)N_NODES * CH * 4);
    float*          xb     = (float*)carve((size_t)N_NODES * CH * 4);
    float*          fin    = (float*)carve((size_t)N_NODES * CH * 4);
    __hip_bfloat16* finb   = (__hip_bfloat16*)carve((size_t)N_NODES * CH * 2);
    __hip_bfloat16* Ab     = (__hip_bfloat16*)carve((size_t)BATCH * CH * 2);
    unsigned char*  mask   = (unsigned char*)carve((size_t)BATCH * ITEM_NUM);

    const long long nf4 = (long long)N_NODES * CH / 4;

    // degrees (int) + normalization
    k_zero_v4<<<(N_NODES / 4 + TPB - 1) / TPB, TPB, 0, stream>>>((v4u*)degu, N_NODES / 4);
    k_deg_u32<<<(N_EDGES + TPB - 1) / TPB, TPB, 0, stream>>>(col, degu, N_EDGES);
    k_dinv<<<(N_NODES + TPB - 1) / TPB, TPB, 0, stream>>>(degu, dinv, N_NODES);

    // CSR build: scan degrees -> start, fill rows
    k_blocksum<<<NB_SCAN, TPB, 0, stream>>>(degu, bsum, N_NODES);
    k_scan_partials<<<1, TPB, 0, stream>>>(bsum, NB_SCAN);
    k_scan_final<<<NB_SCAN, TPB, 0, stream>>>(degu, bsum, startp, N_NODES);
    k_copy_v4<<<(N_NODES / 4 + TPB - 1) / TPB, TPB, 0, stream>>>((v4u*)cursor,
                                                                 (const v4u*)startp, N_NODES / 4);
    k_fill<<<(N_EDGES + TPB - 1) / TPB, TPB, 0, stream>>>(row, col, cursor, csr, N_EDGES);

    // final_x = id_emb
    k_copy_v4<<<(int)((nf4 + TPB - 1) / TPB), TPB, 0, stream>>>((v4u*)fin, (const v4u*)id_emb, nf4);

    // 3 pull-style propagation layers (ping-pong xa/xb), fin += layer output fused
    const float* xc = id_emb;
    float* xn = xa;
    float* xo = xb;
    long long tot = (long long)N_NODES * 16;
    for (int l = 0; l < 3; ++l) {
        k_conv_gather<<<(int)((tot + TPB - 1) / TPB), TPB, 0, stream>>>(xc, startp, degu, csr,
                                                                        dinv, xn, fin, N_NODES);
        xc = xn;
        float* t = xn; xn = xo; xo = t;
    }

    // final_x/4 -> bf16; gather user rows
    long long nel = (long long)N_NODES * CH;
    k_scale_cvt<<<(int)((nel + TPB - 1) / TPB), TPB, 0, stream>>>(fin, finb, nel);
    k_gatherA<<<(BATCH * CH + TPB - 1) / TPB, TPB, 0, stream>>>(finb, uidx, Ab, BATCH * CH);

    // mask init + scatter; zero loss accumulator
    long long m4 = (long long)BATCH * ITEM_NUM / 16;
    k_zero_v4<<<(int)((m4 + TPB - 1) / TPB), TPB, 0, stream>>>((v4u*)mask, m4);
    k_zero1<<<1, 1, 0, stream>>>(out);
    k_scatter<<<(BATCH * SEQ_LEN + TPB - 1) / TPB, TPB, 0, stream>>>(seq, mask, BATCH * SEQ_LEN);

    // WMMA GEMM
    dim3 gg((NTILES + 31) / 32, BATCH / 16);
    k_gemm<<<gg, 256, 0, stream>>>((const __bf16*)Ab,
                                   (const __bf16*)(finb + (size_t)USER_NUM * CH), logits);

    // loss
    k_loss<<<BATCH, 256, 0, stream>>>(logits, mask, out);
}
